// MultiHeadSelfAttention_63333587747411
// MI455X (gfx1250) — matrix operand order repaired and checked by reading, compiled, not run on hardware
//
#include <hip/hip_runtime.h>

typedef __attribute__((ext_vector_type(16))) _Float16 v16h;
typedef __attribute__((ext_vector_type(8)))  _Float16 v8h;
typedef __attribute__((ext_vector_type(8)))  float    v8f;
typedef __attribute__((ext_vector_type(4)))  float    v4f;
typedef __attribute__((ext_vector_type(4)))  unsigned int u32x4;
typedef __attribute__((ext_vector_type(8)))  int      i32x8;
typedef __attribute__((ext_vector_type(4)))  int      i32x4;

union Frag16 { v16h v; v8h h[2]; };

#if __has_builtin(__builtin_amdgcn_tensor_load_to_lds) && __has_builtin(__builtin_amdgcn_s_wait_tensorcnt)
#define HAVE_TDM 1
#else
#define HAVE_TDM 0
#endif

__device__ __forceinline__ v8f wmma_f16(v16h a, v16h b, v8f c) {
    // D = A(16x32 f16) * B(32x16 f16) + C(16x16 f32)
    return __builtin_amdgcn_wmma_f32_16x16x32_f16(false, a, false, b, (short)0, c, false, false);
}

// lane-XOR exchange via ds_swizzle (group-of-32 mode: [14:10]=xor, [9:5]=or, [4:0]=and)
template<int MASK>
__device__ __forceinline__ float swz_xor(float x) {
    int i = __builtin_bit_cast(int, x);
    i = __builtin_amdgcn_ds_swizzle(i, (MASK << 10) | 0x1f);
    return __builtin_bit_cast(float, i);
}

#if HAVE_TDM
// Issue a TDM 2-D tile load (f16 elements) global -> LDS. One call per wave.
// D# per CDNA5 ISA ch.8: group0 = {count, lds_addr, global_addr, type=2},
// group1 = {data_size=2B, tensor dims, tile dims, dim0 stride}; groups 2/3 zero (2-D).
__device__ __forceinline__ void tdm_load_2d(unsigned lds_off, const _Float16* gptr,
                                            unsigned tile_w, unsigned tile_h, unsigned stride)
{
    const unsigned long long ga = (unsigned long long)(size_t)gptr;
    u32x4 g0;
    g0[0] = 1u;                                                  // count=1 (valid user D#)
    g0[1] = lds_off;                                             // lds_addr [63:32]
    g0[2] = (unsigned)ga;                                        // global_addr[31:0]
    g0[3] = (unsigned)((ga >> 32) & 0x01FFFFFFu) | (2u << 30);   // addr[56:32] | type=2
    i32x8 g1;
    g1[0] = 0x00010000;                 // workgroup_mask=0, data_size=1 (2 bytes)
    g1[1] = (int)(tile_w << 16);        // tensor_dim0[15:0]  @ bits[63:48]
    g1[2] = (int)(tile_h << 16);        // tensor_dim1[15:0]  @ bits[95:80]
    g1[3] = (int)(tile_w << 16);        // tile_dim0          @ bits[127:112]
    g1[4] = (int)tile_h;                // tile_dim1          @ bits[143:128]
    g1[5] = (int)stride;                // tensor_dim0_stride[31:0]
    g1[6] = 0; g1[7] = 0;
    i32x4 gz = (i32x4)0;
#if defined(__clang_major__) && (__clang_major__ >= 23)
    __builtin_amdgcn_tensor_load_to_lds(g0, g1, gz, gz, (i32x8)0, 0);
#else
    __builtin_amdgcn_tensor_load_to_lds(g0, g1, gz, gz, 0);
#endif
}
#endif

// ---------------------------------------------------------------------------
// Pre-pass 1: f32 -> f16 copy (8 elements/thread)
// ---------------------------------------------------------------------------
__global__ __launch_bounds__(256)
void f32_to_f16_kernel(const float* __restrict__ in, _Float16* __restrict__ out)
{
    const size_t i = (size_t)blockIdx.x * 256 + threadIdx.x;
    v4f a = ((const v4f*)in)[2 * i], b = ((const v4f*)in)[2 * i + 1];
    v8h o;
#pragma unroll
    for (int j = 0; j < 4; ++j) { o[j] = (_Float16)a[j]; o[4 + j] = (_Float16)b[j]; }
    ((v8h*)out)[i] = o;
}

// ---------------------------------------------------------------------------
// Pre-pass 2: W[K][N] f32 -> Wt[N][K] f16 (tiled 32x32 via LDS)
// ---------------------------------------------------------------------------
__global__ __launch_bounds__(256)
void transpose_f32_to_f16_kernel(const float* __restrict__ in, _Float16* __restrict__ out,
                                 int K, int N)
{
    __shared__ _Float16 tile[32][33];
    const int kBase = blockIdx.y * 32, nBase = blockIdx.x * 32;
    const int c = threadIdx.x & 31, r0 = threadIdx.x >> 5;
#pragma unroll
    for (int rr = 0; rr < 32; rr += 8)
        tile[r0 + rr][c] = (_Float16)in[(size_t)(kBase + r0 + rr) * N + nBase + c];
    __syncthreads();
#pragma unroll
    for (int rr = 0; rr < 32; rr += 8)
        out[(size_t)(nBase + r0 + rr) * K + kBase + c] = tile[c][r0 + rr];
}

// ---------------------------------------------------------------------------
// GEMM: Out[M,N] = A[M,K](f16) * Wt[N,K]^T(f16) + bias[N]
// Block tile 128x256, BK=32, double-buffered LDS, 8 waves of 64x64 (16 WMMA/step).
// Staging: TDM tensor_load_to_lds (wave 0 issues, TENSORcnt-tracked), issued for
// tile t+1 before computing tile t. Fallback: register-staged b128 pipeline.
//   QKVEP: scatter f16: Q,K -> [sel][bh][t][d]; V -> [2][bh][d][t] (pre-transposed,
//          packed b128 stores along t). Else f32 row-major.
// ---------------------------------------------------------------------------
template<bool QKVEP>
__global__ __launch_bounds__(256)
void gemm_wmma_kernel(const _Float16* __restrict__ A,
                      const _Float16* __restrict__ Wt,
                      const float* __restrict__ bias,
                      void* __restrict__ Out,
                      int M, int N, int K)
{
    __shared__ __align__(64) _Float16 As[2][128][32];   // 16 KB
    __shared__ __align__(64) _Float16 Bs[2][256][32];   // 32 KB

    const int tid  = threadIdx.x;
    const int lane = tid & 31;
    const int wid  = tid >> 5;
    const int wm   = wid & 1;        // 2 wave-rows of 64
    const int wn   = wid >> 1;       // 4 wave-cols of 64
    const int half = lane >> 4;
    const int ln   = lane & 15;

    const int rowBase = blockIdx.y * 128;
    const int colBase = blockIdx.x * 256;

    v8f acc[4][4];
#pragma unroll
    for (int i = 0; i < 4; ++i)
#pragma unroll
        for (int j = 0; j < 4; ++j) acc[i][j] = (v8f)0.0f;

    const int nk = K >> 5;

#if HAVE_TDM
    auto stage = [&](int buf, int k0) {
        if (wid == 0) {
            tdm_load_2d((unsigned)(size_t)&As[buf][0][0],
                        A + (size_t)rowBase * K + k0, 32, 128, (unsigned)K);
            tdm_load_2d((unsigned)(size_t)&Bs[buf][0][0],
                        Wt + (size_t)colBase * K + k0, 32, 256, (unsigned)K);
        }
    };
    stage(0, 0);
    if (wid == 0) __builtin_amdgcn_s_wait_tensorcnt(0);
    __syncthreads();
#else
    // register-staged fallback
    const int ar = tid >> 1, ac = (tid & 1) << 4;         // A: 16 halves/thread
    const int brr = tid >> 2, bcc = (tid & 3) << 3;       // B: 4 rows x 8 halves/thread
    auto loadA = [&](int k0, v8h& a0, v8h& a1) {
        const _Float16* src = A + (size_t)(rowBase + ar) * K + k0 + ac;
        a0 = *(const v8h*)src; a1 = *(const v8h*)(src + 8);
    };
    auto loadB = [&](int k0, v8h* g) {
#pragma unroll
        for (int p = 0; p < 4; ++p)
            g[p] = *(const v8h*)(Wt + (size_t)(colBase + p * 64 + brr) * K + k0 + bcc);
    };
    auto storeA = [&](int buf, v8h a0, v8h a1) {
        *(v8h*)&As[buf][ar][ac] = a0; *(v8h*)&As[buf][ar][ac + 8] = a1;
    };
    auto storeB = [&](int buf, const v8h* g) {
#pragma unroll
        for (int p = 0; p < 4; ++p)
            *(v8h*)&Bs[buf][p * 64 + brr][bcc] = g[p];
    };
    v8h ga0, ga1, gb[4];
    loadA(0, ga0, ga1); loadB(0, gb);
    storeA(0, ga0, ga1); storeB(0, gb);
    __syncthreads();
#endif

    for (int kt = 0; kt < nk; ++kt) {
        const int buf = kt & 1;
        const bool more = (kt + 1 < nk);
#if HAVE_TDM
        if (more) stage(buf ^ 1, (kt + 1) << 5);   // DMA next tile under compute
#else
        if (more) { loadA((kt + 1) << 5, ga0, ga1); loadB((kt + 1) << 5, gb); }
#endif

        Frag16 bf[4];
#pragma unroll
        for (int j = 0; j < 4; ++j) {
            const int ncol = wn * 64 + j * 16 + ln;
            bf[j].v = *(const v16h*)&Bs[buf][ncol][16 * half];
        }
#pragma unroll
        for (int i = 0; i < 4; ++i) {
            const int mrow = wm * 64 + i * 16 + ln;
            Frag16 af;
            af.h[0] = *(const v8h*)&As[buf][mrow][16 * half];
            af.h[1] = *(const v8h*)&As[buf][mrow][16 * half + 8];
#pragma unroll
            for (int j = 0; j < 4; ++j)
                acc[i][j] = wmma_f16(af.v, bf[j].v, acc[i][j]);
        }

#if HAVE_TDM
        if (more && wid == 0) __builtin_amdgcn_s_wait_tensorcnt(0);
#else
        if (more) { storeA(buf ^ 1, ga0, ga1); storeB(buf ^ 1, gb); }
#endif
        __syncthreads();
    }

    // epilogue
#pragma unroll
    for (int j = 0; j < 4; ++j) {
        const int col = colBase + wn * 64 + j * 16 + ln;
        const float bv = bias[col];
        const int sel = col >> 10;           // uniform within a 16-wide j-tile
#pragma unroll
        for (int i = 0; i < 4; ++i) {
            if (QKVEP && sel == 2) {
                // V: pre-transposed [bh][d][t]; 8 consecutive t -> one b128 store
                const int rem = col & 1023;
                const int hh = rem >> 6, d = rem & 63;
                const int rowt = rowBase + wm * 64 + i * 16 + 8 * half;  // t of v=0
                const int b_ = rowt >> 11, t0 = rowt & 2047;
                v8h pk;
#pragma unroll
                for (int v = 0; v < 8; ++v) pk[v] = (_Float16)(acc[i][j][v] + bv);
                *(v8h*)((_Float16*)Out +
                        ((((size_t)2 * 2 + b_) * 16 + hh) * 64 + d) * 2048 + t0) = pk;
            } else {
#pragma unroll
                for (int v = 0; v < 8; ++v) {
                    const int row = rowBase + wm * 64 + i * 16 + v + 8 * half;
                    const float val = acc[i][j][v] + bv;
                    if (QKVEP) {
                        const int rem = col & 1023;
                        const int hh = rem >> 6, d = rem & 63;
                        const int b_ = row >> 11, t = row & 2047;
                        ((_Float16*)Out)[((((size_t)sel * 2 + b_) * 16 + hh) * 2048 + t) * 64 + d] =
                            (_Float16)val;
                    } else {
                        ((float*)Out)[(size_t)row * N + col] = val;
                    }
                }
            }
        }
    }
}

// ---------------------------------------------------------------------------
// Flash attention: one workgroup per (b*h, 128-query block); wave owns 16 rows.
// q,k layout: [sel][bh][t][d]; v layout: [2][bh][d][t] (pre-transposed).
// 64-key blocks: 8 QK^T + 8 PV WMMAs per staging round; K/V tiles DMA'd by TDM.
// ---------------------------------------------------------------------------
__global__ __launch_bounds__(256)
void flash_attn_kernel(const _Float16* __restrict__ qkv, _Float16* __restrict__ attOut)
{
    __shared__ __align__(64) _Float16 Kls[64][64];       // [key][d]   8 KB
    __shared__ __align__(64) _Float16 Vt[64][64];        // [d][key]   8 KB
    __shared__ __align__(64) _Float16 Pls[8][16][64];    // per-wave P 16 KB

    const int tid  = threadIdx.x;
    const int lane = tid & 31;
    const int w    = tid >> 5;
    const int half = lane >> 4;
    const int ln   = lane & 15;

    const int bh = blockIdx.y;               // 0..31
    const int b = bh >> 4, h = bh & 15;
    const int qblock = blockIdx.x;
    const int qbase = qblock * 128 + w * 16;

    const size_t strideBH = (size_t)2048 * 64;
    const _Float16* Qg  = qkv + (size_t)bh * strideBH;
    const _Float16* Kg  = qkv + (size_t)(32 + bh) * strideBH;
    const _Float16* Vtg = qkv + (size_t)(64 + bh) * strideBH;   // [d][t]

    // Q fragments (two 16x32 k-steps over D=64), held in registers
    Frag16 qf[2];
    {
        const _Float16* qrow = Qg + (size_t)(qbase + ln) * 64;
#pragma unroll
        for (int s = 0; s < 2; ++s) {
            qf[s].h[0] = *(const v8h*)(qrow + 32 * s + 16 * half);
            qf[s].h[1] = *(const v8h*)(qrow + 32 * s + 16 * half + 8);
        }
    }

    float m_run[8], l_run[8];
    v8f o[4];
#pragma unroll
    for (int v = 0; v < 8; ++v) { m_run[v] = -1e30f; l_run[v] = 0.0f; }
#pragma unroll
    for (int j = 0; j < 4; ++j) o[j] = (v8f)0.0f;

#if !HAVE_TDM
    const int sr = tid >> 2;            // 0..63
    const int sc = (tid & 3) << 4;      // 0,16,32,48
#endif

    const int nkb = qblock * 2 + 2;     // causal: 64-key blocks <= query block
    for (int kb = 0; kb < nkb; ++kb) {
        const int kbase = kb * 64;
        __syncthreads();
        // ---- stage K [key][d] and Vt [d][key] ----
#if HAVE_TDM
        if (w == 0) {
            tdm_load_2d((unsigned)(size_t)&Kls[0][0], Kg + (size_t)kbase * 64, 64, 64, 64);
            tdm_load_2d((unsigned)(size_t)&Vt[0][0],  Vtg + kbase,             64, 64, 2048);
            __builtin_amdgcn_s_wait_tensorcnt(0);
        }
#else
        {
            const _Float16* ks = Kg + (size_t)(kbase + sr) * 64 + sc;
            v8h k0 = *(const v8h*)ks, k1 = *(const v8h*)(ks + 8);
            const _Float16* vs = Vtg + (size_t)sr * 2048 + kbase + sc;
            v8h v0 = *(const v8h*)vs, v1 = *(const v8h*)(vs + 8);
            *(v8h*)&Kls[sr][sc]     = k0;  *(v8h*)&Kls[sr][sc + 8] = k1;
            *(v8h*)&Vt[sr][sc]      = v0;  *(v8h*)&Vt[sr][sc + 8]  = v1;
        }
#endif
        __syncthreads();

        // ---- S = Q * K^T  (16 x 64 = four 16x16 tiles) ----
        v8f s[4];
#pragma unroll
        for (int j = 0; j < 4; ++j) s[j] = (v8f)0.0f;
#pragma unroll
        for (int ks = 0; ks < 2; ++ks)
#pragma unroll
            for (int j = 0; j < 4; ++j) {
                v16h bm = *(const v16h*)&Kls[j * 16 + ln][32 * ks + 16 * half];
                s[j] = wmma_f16(qf[ks].v, bm, s[j]);
            }

        // causal masking only needed on/near the diagonal (wave-uniform branch)
        const bool needMask = (kbase + 63) > qbase;

        // ---- online softmax (rows live on 16 lanes of each half) ----
        float p[4][8], alpha[8];
#pragma unroll
        for (int v = 0; v < 8; ++v) {
            const int qidx = qbase + v + 8 * half;
            float sv[4];
#pragma unroll
            for (int j = 0; j < 4; ++j) {
                sv[j] = s[j][v] * 0.125f;
                if (needMask && (kbase + j * 16 + ln > qidx)) sv[j] = -1e30f;
            }
            float rm = fmaxf(fmaxf(sv[0], sv[1]), fmaxf(sv[2], sv[3]));
            rm = fmaxf(rm, swz_xor<1>(rm));
            rm = fmaxf(rm, swz_xor<2>(rm));
            rm = fmaxf(rm, swz_xor<4>(rm));
            rm = fmaxf(rm, swz_xor<8>(rm));
            const float nm = fmaxf(m_run[v], rm);
            float rs = 0.0f;
#pragma unroll
            for (int j = 0; j < 4; ++j) { p[j][v] = __expf(sv[j] - nm); rs += p[j][v]; }
            rs += swz_xor<1>(rs);
            rs += swz_xor<2>(rs);
            rs += swz_xor<4>(rs);
            rs += swz_xor<8>(rs);
            alpha[v] = __expf(m_run[v] - nm);
            l_run[v] = l_run[v] * alpha[v] + rs;
            m_run[v] = nm;
        }
#pragma unroll
        for (int j = 0; j < 4; ++j)
#pragma unroll
            for (int v = 0; v < 8; ++v) o[j][v] *= alpha[v];

        // ---- P: C-layout -> LDS -> A-layout fragments (16x64 = 2 k-steps) ----
#pragma unroll
        for (int v = 0; v < 8; ++v)
#pragma unroll
            for (int j = 0; j < 4; ++j)
                Pls[w][v + 8 * half][j * 16 + ln] = (_Float16)p[j][v];
        Frag16 pf[2];
#pragma unroll
        for (int ks = 0; ks < 2; ++ks) {
            pf[ks].h[0] = *(const v8h*)&Pls[w][ln][32 * ks + 16 * half];
            pf[ks].h[1] = *(const v8h*)&Pls[w][ln][32 * ks + 16 * half + 8];
        }

        // ---- O += P * V ----
#pragma unroll
        for (int ks = 0; ks < 2; ++ks)
#pragma unroll
            for (int j = 0; j < 4; ++j) {
                v16h bm = *(const v16h*)&Vt[j * 16 + ln][32 * ks + 16 * half];
                o[j] = wmma_f16(pf[ks].v, bm, o[j]);
            }
    }

    // ---- normalize + store f16 at [b][t][h*64+d] ----
    float inv[8];
#pragma unroll
    for (int v = 0; v < 8; ++v) inv[v] = 1.0f / l_run[v];
#pragma unroll
    for (int j = 0; j < 4; ++j) {
#pragma unroll
        for (int v = 0; v < 8; ++v) {
            const int t = qbase + v + 8 * half;
            const int d = j * 16 + ln;
            attOut[((size_t)(b * 2048 + t) * 16 + h) * 64 + d] = (_Float16)(o[j][v] * inv[v]);
        }
    }
}

// ---------------------------------------------------------------------------
extern "C" void kernel_launch(void* const* d_in, const int* in_sizes, int n_in,
                              void* d_out, int out_size, void* d_ws, size_t ws_size,
                              hipStream_t stream)
{
    const float* x     = (const float*)d_in[0];
    const float* Wqkv  = (const float*)d_in[1];
    const float* bqkv  = (const float*)d_in[2];
    const float* Wproj = (const float*)d_in[3];
    const float* bproj = (const float*)d_in[4];

    // workspace layout (f16 elements)
    _Float16* qkv    = (_Float16*)d_ws;                          // 3*2*16*2048*64 = 12.58M
    _Float16* attOut = qkv    + (size_t)3 * 2 * 16 * 2048 * 64;  // 4096*1024
    _Float16* xh     = attOut + (size_t)4096 * 1024;             // 4096*1024
    _Float16* WqkvT  = xh     + (size_t)4096 * 1024;             // 3072*1024
    _Float16* WprojT = WqkvT  + (size_t)3072 * 1024;             // 1024*1024

    dim3 blk(256);
    // pre-passes: one-shot f16 conversion / weight transposition (L2-resident)
    f32_to_f16_kernel<<<dim3((4096 * 1024) / (256 * 8)), blk, 0, stream>>>(x, xh);
    transpose_f32_to_f16_kernel<<<dim3(3072 / 32, 1024 / 32), blk, 0, stream>>>(Wqkv, WqkvT, 1024, 3072);
    transpose_f32_to_f16_kernel<<<dim3(1024 / 32, 1024 / 32), blk, 0, stream>>>(Wproj, WprojT, 1024, 1024);

    // 1) qkv = x @ Wqkv + bqkv -> Q,K:[sel][bh][t][d], V:[2][bh][d][t] (f16)
    gemm_wmma_kernel<true><<<dim3(3072 / 256, 4096 / 128), blk, 0, stream>>>(
        xh, WqkvT, bqkv, (void*)qkv, 4096, 3072, 1024);
    // 2) causal flash attention per head
    flash_attn_kernel<<<dim3(2048 / 128, 32), blk, 0, stream>>>(qkv, attOut);
    // 3) out = attOut @ Wproj + bproj (f32)
    gemm_wmma_kernel<false><<<dim3(1024 / 256, 4096 / 128), blk, 0, stream>>>(
        attOut, WprojT, bproj, d_out, 4096, 1024, 1024);
}
